// FastDTWEmbeddingSimilarity_57741540327823
// MI455X (gfx1250) — matrix operand order repaired and checked by reading, not compiled, not run on hardware
//
#include <hip/hip_runtime.h>

// ---------------------------------------------------------------------------
// FastDTW over cosine distances, MI455X (gfx1250, wave32, WMMA bf16).
//   B=256 batches, N=M=256, D=1024, band radius = 128, BIG = 1e30.
// Pipeline:
//   k1: row inverse-norms for seq1/seq2            (1 wave32 per row)
//   k2: batched GEMM -> dist = 1 - clamp(cos,-1,1) (v_wmma_f32_16x16x32_bf16,
//       software-pipelined global->LDS staging, clustered B-fragment loads)
//   k3: banded DTW via wave32 min-plus scans       (1 wave per batch)
// ---------------------------------------------------------------------------

typedef __attribute__((ext_vector_type(16))) __bf16 v16bf;
typedef __attribute__((ext_vector_type(8)))  __bf16 v8bf;
typedef __attribute__((ext_vector_type(8)))  float  v8f;

#define BATCH   256
#define SEQN    256
#define DIMD    1024
#define RADIUS  128
#define BIGF    1e30f

// ---------------------------------------------------------------- norms ----
__global__ __launch_bounds__(256) void norm_kernel(const float* __restrict__ s1,
                                                   const float* __restrict__ s2,
                                                   float* __restrict__ inv1,
                                                   float* __restrict__ inv2) {
  const int gw   = (blockIdx.x * blockDim.x + threadIdx.x) >> 5;  // global wave = row id
  const int lane = threadIdx.x & 31;
  const float* src;
  float* dst;
  int row;
  if (gw < BATCH * SEQN) { src = s1; dst = inv1; row = gw; }
  else                   { src = s2; dst = inv2; row = gw - BATCH * SEQN; }

  const float4* q = (const float4*)(src + (size_t)row * DIMD + lane * 32);
  float s = 0.f;
#pragma unroll
  for (int t = 0; t < 8; ++t) {
    float4 v = q[t];
    s += v.x * v.x + v.y * v.y + v.z * v.z + v.w * v.w;
  }
#pragma unroll
  for (int off = 16; off > 0; off >>= 1) s += __shfl_xor(s, off, 32);
  if (lane == 0) dst[row] = 1.0f / fmaxf(sqrtf(s), 1e-12f);
}

// ------------------------------------------------------- GEMM -> dist ------
// Block: 256 threads (8 waves). Block tile: 128(M) x 128(N). K-chunk: 32.
// Wave w covers M rows [16w, 16w+16) x all 128 N cols  => 8 wmma tiles.
// Two-stage pipeline: chunk c is cvt+stored to LDS while chunk c+1's global
// loads are issued before the barrier, hiding HBM latency behind WMMA work.
#define TM 128
#define TN 128
#define KC 32
#define NCHUNK (DIMD / KC)
#define LPAD 8   // LDS row pad (elements) -> 80B row stride, 16B aligned chunks

__global__ __launch_bounds__(256) void gemm_dist_kernel(const float* __restrict__ X,
                                                        const float* __restrict__ Y,
                                                        const float* __restrict__ invn1,
                                                        const float* __restrict__ invn2,
                                                        float* __restrict__ dist) {
  const int b    = blockIdx.y;
  const int bi0  = (blockIdx.x & 1) * TM;
  const int bj0  = (blockIdx.x >> 1) * TN;
  const int tid  = threadIdx.x;
  const int lane = tid & 31;
  const int wave = tid >> 5;

  __shared__ __bf16 Xs[TM][KC + LPAD];
  __shared__ __bf16 Ys[TN][KC + LPAD];

  const float* Xb = X + ((size_t)b * SEQN + bi0) * DIMD;
  const float* Yb = Y + ((size_t)b * SEQN + bj0) * DIMD;

  v8f acc[8];
  const v8f zero = {0.f, 0.f, 0.f, 0.f, 0.f, 0.f, 0.f, 0.f};
#pragma unroll
  for (int t = 0; t < 8; ++t) acc[t] = zero;

  // staging role: thread -> (row = tid/2, 16-element half of the K-chunk)
  const int srow  = tid >> 1;
  const int shalf = (tid & 1) * 16;
  const float* xrow = Xb + (size_t)srow * DIMD + shalf;
  const float* yrow = Yb + (size_t)srow * DIMD + shalf;

  // fragment addresses (constant across chunks)
  const int   mrow = wave * 16 + (lane & 15);
  const int   aoff = (lane >= 16) ? 8 : 0;
  const int   boff = (lane >= 16) ? 8 : 0;

  float4 cx[4], cy[4];   // current chunk staging registers
#pragma unroll
  for (int q = 0; q < 4; ++q) {
    cx[q] = ((const float4*)xrow)[q];
    cy[q] = ((const float4*)yrow)[q];
  }

  for (int c = 0; c < NCHUNK; ++c) {
    // ---- cvt + store current chunk to LDS ----
#pragma unroll
    for (int q = 0; q < 4; ++q) {
      Xs[srow][shalf + 4 * q + 0] = (__bf16)cx[q].x;
      Xs[srow][shalf + 4 * q + 1] = (__bf16)cx[q].y;
      Xs[srow][shalf + 4 * q + 2] = (__bf16)cx[q].z;
      Xs[srow][shalf + 4 * q + 3] = (__bf16)cx[q].w;
      Ys[srow][shalf + 4 * q + 0] = (__bf16)cy[q].x;
      Ys[srow][shalf + 4 * q + 1] = (__bf16)cy[q].y;
      Ys[srow][shalf + 4 * q + 2] = (__bf16)cy[q].z;
      Ys[srow][shalf + 4 * q + 3] = (__bf16)cy[q].w;
    }

    // ---- issue next chunk's global loads (in flight across the WMMA phase) ----
    float4 nx[4], ny[4];
    if (c + 1 < NCHUNK) {
      const float* xn = xrow + (c + 1) * KC;
      const float* yn = yrow + (c + 1) * KC;
#pragma unroll
      for (int q = 0; q < 4; ++q) {
        nx[q] = ((const float4*)xn)[q];
        ny[q] = ((const float4*)yn)[q];
      }
    }

    __syncthreads();  // LDS chunk visible

    // A fragment (16x32 bf16, ISA layout): lanes 0-15 hold M=lane, K={0..7,16..23};
    // lanes 16-31 hold M=lane-16, K={8..15,24..31}.
    v8bf alo = *(const v8bf*)&Xs[mrow][aoff];
    v8bf ahi = *(const v8bf*)&Xs[mrow][aoff + 16];
    v16bf afrag;
#pragma unroll
    for (int e = 0; e < 8; ++e) { afrag[e] = alo[e]; afrag[e + 8] = ahi[e]; }

    // B fragments (32x16 bf16): lanes 0-15 hold N=lane, K={0..7,16..23}; lanes 16-31
    // hold N=lane-16, K={8..15,24..31}. Load all 8 tiles first (one DS clause), then
    // run the 8 WMMAs back-to-back.
    v16bf bfrag[8];
#pragma unroll
    for (int t = 0; t < 8; ++t) {
      const int nrow = t * 16 + (lane & 15);
      v8bf blo = *(const v8bf*)&Ys[nrow][boff];
      v8bf bhi = *(const v8bf*)&Ys[nrow][boff + 16];
#pragma unroll
      for (int e = 0; e < 8; ++e) { bfrag[t][e] = blo[e]; bfrag[t][e + 8] = bhi[e]; }
    }
#pragma unroll
    for (int t = 0; t < 8; ++t) {
      acc[t] = __builtin_amdgcn_wmma_f32_16x16x32_bf16(
          false, afrag, false, bfrag[t], (short)0, acc[t], false, false);
    }

    __syncthreads();  // LDS chunk consumed

#pragma unroll
    for (int q = 0; q < 4; ++q) { cx[q] = nx[q]; cy[q] = ny[q]; }
  }

  // Epilogue: C/D layout — VGPR r: lanes 0-15 = (M=r, N=lane); lanes 16-31 = (M=8+r, N=lane-16)
  const int jloc = lane & 15;
  const int moff = (lane >= 16) ? 8 : 0;
  float in1[8];
#pragma unroll
  for (int r = 0; r < 8; ++r) in1[r] = invn1[b * SEQN + bi0 + wave * 16 + moff + r];
#pragma unroll
  for (int t = 0; t < 8; ++t) {
    const int j = bj0 + t * 16 + jloc;
    const float iny = invn2[b * SEQN + j];
#pragma unroll
    for (int r = 0; r < 8; ++r) {
      const int i = bi0 + wave * 16 + moff + r;
      float s = acc[t][r] * in1[r] * iny;
      s = fminf(fmaxf(s, -1.f), 1.f);
      dist[((size_t)b * SEQN + i) * SEQN + j] = 1.f - s;
    }
  }
}

// ------------------------------------------------------------- DTW ---------
// One wave32 per batch. Lane L owns columns 8L..8L+7 in registers.
// Row recurrence solved with the same min-plus scan `comb` as the reference:
//   comb((sl,tl),(sr,tr)) = (sl+sr, min(tr, tl+sr))
__global__ __launch_bounds__(256) void dtw_kernel(const float* __restrict__ dist,
                                                  float* __restrict__ out) {
  const int bidx = (blockIdx.x * blockDim.x + threadIdx.x) >> 5;  // batch
  const int lane = threadIdx.x & 31;
  const float* Db = dist + (size_t)bidx * SEQN * SEQN;

  float prev[8];

  // ---- row 0: masked cumulative sum (j <= RADIUS else BIG) ----
  {
    const float* rp = Db + lane * 8;
    float4 d0 = ((const float4*)rp)[0];
    float4 d1 = ((const float4*)rp)[1];
    float d[8] = {d0.x, d0.y, d0.z, d0.w, d1.x, d1.y, d1.z, d1.w};
    float run = 0.f, pref[8];
#pragma unroll
    for (int e = 0; e < 8; ++e) { run += d[e]; pref[e] = run; }
    float sc = run;
#pragma unroll
    for (int off = 1; off < 32; off <<= 1) {
      float s = __shfl_up(sc, off, 32);
      if (lane >= off) sc += s;
    }
    float excl = __shfl_up(sc, 1, 32);
    if (lane == 0) excl = 0.f;
#pragma unroll
    for (int e = 0; e < 8; ++e) {
      const int j = 8 * lane + e;
      prev[e] = (j <= RADIUS) ? (excl + pref[e]) : BIGF;
    }
  }

  // ---- rows 1..255 ----
  for (int i = 1; i < SEQN; ++i) {
    const float* rp = Db + (size_t)i * SEQN + lane * 8;
    if (i + 1 < SEQN) __builtin_prefetch(rp + SEQN, 0, 0);  // global_prefetch next row
    float4 d0 = ((const float4*)rp)[0];
    float4 d1 = ((const float4*)rp)[1];
    float d[8] = {d0.x, d0.y, d0.z, d0.w, d1.x, d1.y, d1.z, d1.w};

    const float prevleft = __shfl_up(prev[7], 1, 32);  // acc[i-1][8L-1]
    float a0 = (i <= RADIUS) ? (prev[0] + d[0]) : BIGF;  // column 0 (valid on lane 0)
    a0 = __shfl(a0, 0, 32);

    float st[8], bt[8];
    float pm1 = prevleft;
#pragma unroll
    for (int e = 0; e < 8; ++e) {
      const int j = 8 * lane + e;
      const float c = fminf(prev[e], pm1);  // min(acc[i-1][j], acc[i-1][j-1])
      const bool inb = (j >= 1) && (j >= i - RADIUS) && (j <= i + RADIUS);
      bt[e] = inb ? (c + d[e]) : BIGF;
      st[e] = inb ? d[e] : BIGF;
      pm1 = prev[e];
    }
    if (lane == 0) { st[0] = 0.f; bt[0] = BIGF; }  // column 0 excluded: identity elem

    // lane-local inclusive min-plus compose
    float S = 0.f, T = BIGF, Sl[8], Tl[8];
#pragma unroll
    for (int e = 0; e < 8; ++e) {
      T = fminf(bt[e], T + st[e]);
      S = S + st[e];
      Sl[e] = S; Tl[e] = T;
    }

    // exclusive inter-lane min-plus scan of (S,T)
    float Si = S, Ti = T;
#pragma unroll
    for (int off = 1; off < 32; off <<= 1) {
      float Ss = __shfl_up(Si, off, 32);
      float Ts = __shfl_up(Ti, off, 32);
      if (lane >= off) { Ti = fminf(Ti, Ts + Si); Si = Ss + Si; }
    }
    float Se = __shfl_up(Si, 1, 32);
    float Te = __shfl_up(Ti, 1, 32);
    if (lane == 0) { Se = 0.f; Te = BIGF; }

    // apply prefixes + boundary a0: row[j] = min(T_j, a0 + S_j)
#pragma unroll
    for (int e = 0; e < 8; ++e) {
      const float Sp = Se + Sl[e];
      const float Tp = fminf(Tl[e], Te + Sl[e]);
      const int j = 8 * lane + e;
      const float v = fminf(Tp, a0 + Sp);
      prev[e] = (j == 0) ? a0 : v;
    }
  }

  if (lane == 31) out[bidx] = prev[7];  // acc[255][255]
}

// ---------------------------------------------------------------------------
extern "C" void kernel_launch(void* const* d_in, const int* in_sizes, int n_in,
                              void* d_out, int out_size, void* d_ws, size_t ws_size,
                              hipStream_t stream) {
  const float* seq1 = (const float*)d_in[0];
  const float* seq2 = (const float*)d_in[1];
  float* out = (float*)d_out;

  char* ws = (char*)d_ws;
  float* invn1 = (float*)(ws);                                  // 256 KB
  float* invn2 = (float*)(ws + (size_t)BATCH * SEQN * 4);       // 256 KB
  float* dist  = (float*)(ws + (size_t)2 * BATCH * SEQN * 4);   // 64 MB

  // 1) inverse norms: 2*B*N rows, one wave per row, 8 waves per block
  const int nrows = 2 * BATCH * SEQN;
  norm_kernel<<<nrows / 8, 256, 0, stream>>>(seq1, seq2, invn1, invn2);

  // 2) batched GEMM -> distance matrices (WMMA bf16, f32 accumulate)
  dim3 g2(4, BATCH);  // 4 tiles of 128x128 per batch
  gemm_dist_kernel<<<g2, 256, 0, stream>>>(seq1, seq2, invn1, invn2, dist);

  // 3) banded DTW, one wave32 per batch
  dtw_kernel<<<BATCH / 8, 256, 0, stream>>>(dist, out);
}
